// PiKVCompressor_22170621182521
// MI455X (gfx1250) — matrix operand order repaired and checked by reading, not compiled, not run on hardware
//
#include <hip/hip_runtime.h>
#include <hip/hip_bf16.h>

typedef _Float16 half16 __attribute__((ext_vector_type(16)));
typedef _Float16 half8  __attribute__((ext_vector_type(8)));
typedef float    float8 __attribute__((ext_vector_type(8)));

#define NWAVES 16          // 512 threads / block
#define MROWS  64          // tokens per block (4 M-tiles of 16)
#define EPS 1e-5f

// Real dims: 1024 -> 512 -> 204 -> 65 ; padded (K mult 32): 1024, 512, 224, 96

// ---------------------------------------------------------------------------
// Weight pack: fp32 (Nr x Kr) row-major -> f16 swizzled WMMA-B tiles, padded.
// tile t covers kb = (t / (Np/16))*32, nb = (t % (Np/16))*16 ; 512 halves/tile
// lane L holds B col n = nb+L%16 : halves j = W[n][kb+(j/8)*16+(L/16)*8+j%8]
// ---------------------------------------------------------------------------
__global__ void pikv_pack_w(const float* __restrict__ src, _Float16* __restrict__ dst,
                            int Nr, int Kr, int Np, int Kp) {
    int idx = blockIdx.x * 256 + threadIdx.x;
    int total = Np * Kp;
    if (idx >= total) return;
    int t = idx >> 9;
    int r = idx & 511;
    int lane = r >> 4, j = r & 15;
    int ntiles = Np >> 4;
    int n = (t % ntiles) * 16 + (lane & 15);
    int k = (t / ntiles) * 32 + ((j & 8) << 1) + ((lane >> 4) << 3) + (j & 7);
    float v = (n < Nr && k < Kr) ? src[(size_t)n * Kr + k] : 0.0f;
    dst[idx] = (_Float16)v;
}

// ---------------------------------------------------------------------------
// 64xN GEMM: actLDS (64 x Kp f16 row-major) @ packed W (Kp x Np) -> acc f32.
// Four 16-row M-tiles share each B fragment in-register (4 WMMAs per B load).
// A-fragment per ISA 16-bit A 16x32 layout:
//   lane: m = L%16 ; halves 0..7 = K[kb+(L/16)*8+j], 8..15 = K[kb+16+(L/16)*8+j]
// ---------------------------------------------------------------------------
__device__ __forceinline__ half16 load_a(const _Float16* __restrict__ act, int Kp,
                                         int m, int kb, int off) {
    half8 lo = *(const half8*)(act + (size_t)m * Kp + kb + off);
    half8 hi = *(const half8*)(act + (size_t)m * Kp + kb + 16 + off);
    half16 a;
#pragma unroll
    for (int j = 0; j < 8; ++j) { a[j] = lo[j]; a[8 + j] = hi[j]; }
    return a;
}

__device__ __forceinline__ void gemm_tile4(const _Float16* __restrict__ actLDS, int Kp,
                                           const _Float16* __restrict__ wpack, int Np,
                                           float* __restrict__ accLDS, int lane, int wave) {
    const int ntiles = Np >> 4;
    const int m    = lane & 15;
    const int off  = (lane >> 4) << 3;
    const int row0 = (lane >> 4) << 3;
    for (int nt = wave; nt < ntiles; nt += NWAVES) {
        float8 c0 = {}, c1 = {}, c2 = {}, c3 = {};
        for (int kb = 0; kb < Kp; kb += 32) {
            half16 b = *((const half16*)(wpack + ((size_t)(kb >> 5) * ntiles + nt) * 512) + lane);
            half16 a0 = load_a(actLDS, Kp, m,      kb, off);
            half16 a1 = load_a(actLDS, Kp, m + 16, kb, off);
            half16 a2 = load_a(actLDS, Kp, m + 32, kb, off);
            half16 a3 = load_a(actLDS, Kp, m + 48, kb, off);
            c0 = __builtin_amdgcn_wmma_f32_16x16x32_f16(false, a0, false, b, (short)0, c0, false, false);
            c1 = __builtin_amdgcn_wmma_f32_16x16x32_f16(false, a1, false, b, (short)0, c1, false, false);
            c2 = __builtin_amdgcn_wmma_f32_16x16x32_f16(false, a2, false, b, (short)0, c2, false, false);
            c3 = __builtin_amdgcn_wmma_f32_16x16x32_f16(false, a3, false, b, (short)0, c3, false, false);
        }
        int col = nt * 16 + (lane & 15);
#pragma unroll
        for (int i = 0; i < 8; ++i) {
            accLDS[(size_t)(row0 + i) * Np + col]      = c0[i];
            accLDS[(size_t)(row0 + i + 16) * Np + col] = c1[i];
            accLDS[(size_t)(row0 + i + 32) * Np + col] = c2[i];
            accLDS[(size_t)(row0 + i + 48) * Np + col] = c3[i];
        }
    }
}

// ---------------------------------------------------------------------------
// LayerNorm over Nr real cols of acc (64 x Np f32), + bias pre-LN, optional ReLU,
// write f16 (pad cols zeroed). Optional per-row select: rows with imp >= 0.5
// keep dst's existing contents (dst holds enc2; self-aliased read-then-write
// per (row, n) by the same thread, so this is race-free).
// 16 waves x 2 slots x 2 rows = 64 rows; 16-lane xor-shuffle reduction.
// ---------------------------------------------------------------------------
__device__ __forceinline__ void ln_store(const float* __restrict__ acc, int Np, int Nr,
                                         const float* __restrict__ bias,
                                         const float* __restrict__ g,
                                         const float* __restrict__ beta,
                                         bool relu, _Float16* __restrict__ dst,
                                         int lane, int wave,
                                         const float* __restrict__ imp) {
    int c = lane & 15;
#pragma unroll
    for (int slot = 0; slot < 2; ++slot) {
        int row = 4 * wave + 2 * slot + (lane >> 4);
        float s = 0.f, s2 = 0.f;
        for (int n = c; n < Nr; n += 16) {
            float v = acc[(size_t)row * Np + n] + bias[n];
            s += v; s2 += v * v;
        }
#pragma unroll
        for (int mask = 8; mask >= 1; mask >>= 1) {
            s  += __shfl_xor(s,  mask, 32);
            s2 += __shfl_xor(s2, mask, 32);
        }
        float mu = s / (float)Nr;
        float var = s2 / (float)Nr - mu * mu;
        float rs = rsqrtf(var + EPS);
        bool sel = (imp != nullptr) && (imp[row] >= 0.5f);   // level==1: keep enc2
        for (int n = c; n < Np; n += 16) {
            _Float16 o = (_Float16)0.f;
            if (n < Nr) {
                float v = acc[(size_t)row * Np + n] + bias[n];
                float y = (v - mu) * rs * g[n] + beta[n];
                if (relu) y = fmaxf(y, 0.f);
                o = (_Float16)y;
            }
            if (sel) o = dst[(size_t)row * Np + n];          // self-alias: keep old value
            dst[(size_t)row * Np + n] = o;
        }
    }
}

__device__ __forceinline__ void ln_residual_out(const float* __restrict__ acc, int Np,
                                                const float* __restrict__ bias,
                                                const float* __restrict__ g,
                                                const float* __restrict__ beta,
                                                const float* __restrict__ xin,
                                                float* __restrict__ xout,
                                                int lane, int wave) {
    int c = lane & 15;
#pragma unroll
    for (int slot = 0; slot < 2; ++slot) {
        int row = 4 * wave + 2 * slot + (lane >> 4);
        float s = 0.f, s2 = 0.f;
        for (int n = c; n < Np; n += 16) {
            float v = acc[(size_t)row * Np + n] + bias[n];
            s += v; s2 += v * v;
        }
#pragma unroll
        for (int mask = 8; mask >= 1; mask >>= 1) {
            s  += __shfl_xor(s,  mask, 32);
            s2 += __shfl_xor(s2, mask, 32);
        }
        float mu = s / (float)Np;
        float var = s2 / (float)Np - mu * mu;
        float rs = rsqrtf(var + EPS);
        for (int n = c; n < Np; n += 16) {
            float v = acc[(size_t)row * Np + n] + bias[n];
            float y = (v - mu) * rs * g[n] + beta[n];
            xout[(size_t)row * Np + n] = xin[(size_t)row * Np + n] + y;
        }
    }
}

// ---------------------------------------------------------------------------
struct PikvParams {
    const float *keys, *values, *importance;
    const _Float16 *We0, *We1, *We2, *Wd2, *Wd1, *Wd0;
    const float *eb0, *eb1, *eb2, *eg0, *eg1, *eg2, *et0, *et1, *et2;
    const float *db0, *db1, *db2, *dg0, *dg1, *dg2, *dt0, *dt1, *dt2;
    float* out;
    int tokens;       // tokens per tensor (B*S)
};

// LDS carve (320 KB total; lifetimes overlapped):
//  acc f32 region [0, 262144)      : S1..S5 use <=131072 of it; S6 uses all 262144.
//  upper half of acc region is time-shared:
//    actX f16 64x1024 @131072 (131072B) : S0..S1 gemm input, dead after S1 gemm
//    actC f16 64x224  @131072 (28672B)  : S2 out / S3 in / S4 out / S5 in
//    actD f16 64x96   @159744 (12288B)  : S3 out / S4 in
//  actB f16 64x512 @262144 (65536B)     : S1 out / S2 in ; S5 out / S6 in
#define OFF_ACC   0
#define OFF_ACTX  131072
#define OFF_ACTC  131072
#define OFF_ACTD  159744
#define OFF_ACTB  262144
#define LDS_BYTES (262144 + 65536)

__global__ __launch_bounds__(512) void pikv_main(PikvParams p) {
    extern __shared__ char smem[];
    float*     acc  = (float*)(smem + OFF_ACC);
    _Float16*  actX = (_Float16*)(smem + OFF_ACTX);
    _Float16*  actC = (_Float16*)(smem + OFF_ACTC);
    _Float16*  actD = (_Float16*)(smem + OFF_ACTD);
    _Float16*  actB = (_Float16*)(smem + OFF_ACTB);

    const int tid = threadIdx.x;
    const int lane = tid & 31, wave = tid >> 5;
    const int tiles_per_tensor = p.tokens / MROWS;
    const int tensor = blockIdx.x / tiles_per_tensor;
    const int tloc   = blockIdx.x % tiles_per_tensor;

    const float* xin  = (tensor ? p.values : p.keys) + (size_t)tloc * MROWS * 1024;
    float*       xout = p.out + (size_t)tensor * p.tokens * 1024 + (size_t)tloc * MROWS * 1024;
    const float* impT = p.importance + (size_t)tloc * MROWS;   // shared by keys & values

    // S0: x -> f16
    for (int i = tid; i < MROWS * 1024; i += 512) actX[i] = (_Float16)xin[i];
    __syncthreads();

    // S1: enc0  1024 -> 512, LN+ReLU          (actX dead after this gemm)
    gemm_tile4(actX, 1024, p.We0, 512, acc, lane, wave);
    __syncthreads();
    ln_store(acc, 512, 512, p.eb0, p.eg0, p.et0, true, actB, lane, wave, nullptr);
    __syncthreads();

    // S2: enc1  512 -> 204(224), LN+ReLU      (actC = enc2, kept through S4)
    gemm_tile4(actB, 512, p.We1, 224, acc, lane, wave);
    __syncthreads();
    ln_store(acc, 224, 204, p.eb1, p.eg1, p.et1, true, actC, lane, wave, nullptr);
    __syncthreads();

    // S3: enc2  204(224) -> 65(96), LN+ReLU
    gemm_tile4(actC, 224, p.We2, 96, acc, lane, wave);
    __syncthreads();
    ln_store(acc, 96, 65, p.eb2, p.eg2, p.et2, true, actD, lane, wave, nullptr);
    __syncthreads();

    // S4: dec2  65(96) -> 204(224), LN ; rows with imp>=0.5 (level 1) keep enc2
    //     (threshold quirk in reference makes level 0 unreachable)
    gemm_tile4(actD, 96, p.Wd2, 224, acc, lane, wave);
    __syncthreads();
    ln_store(acc, 224, 204, p.db2, p.dg2, p.dt2, false, actC, lane, wave, impT);
    __syncthreads();

    // S5: dec1  204(224) -> 512, LN
    gemm_tile4(actC, 224, p.Wd1, 512, acc, lane, wave);
    __syncthreads();
    ln_store(acc, 512, 512, p.db1, p.dg1, p.dt1, false, actB, lane, wave, nullptr);
    __syncthreads();

    // S6: dec0  512 -> 1024 (acc grows to full 256KB; actC/actD region dead),
    //     LN, residual add, f32 out
    gemm_tile4(actB, 512, p.Wd0, 1024, acc, lane, wave);
    __syncthreads();
    ln_residual_out(acc, 1024, p.db0, p.dg0, p.dt0, xin, xout, lane, wave);
}

// ---------------------------------------------------------------------------
extern "C" void kernel_launch(void* const* d_in, const int* in_sizes, int n_in,
                              void* d_out, int out_size, void* d_ws, size_t ws_size,
                              hipStream_t stream) {
    (void)n_in; (void)out_size; (void)ws_size;

    // Param index maps: dict-insertion order vs sorted(pytree) order; pick by size.
    // insertion: enc_w@3, enc_b@6, enc_g@9, enc_beta@12, dec_w@15, dec_b@18, dec_g@21, dec_beta@24
    // sorted:    dec_b@3, dec_beta@6, dec_g@9, dec_w@12, enc_b@15, enc_beta@18, enc_g@21, enc_w@24
    int EW, EB, EG, ET, DW, DB, DG, DT;
    if (in_sizes[3] > 10000) { EW = 3; EB = 6; EG = 9; ET = 12; DW = 15; DB = 18; DG = 21; DT = 24; }
    else                     { DB = 3; DT = 6; DG = 9; DW = 12; EB = 15; ET = 18; EG = 21; EW = 24; }

    PikvParams p;
    p.keys       = (const float*)d_in[0];
    p.values     = (const float*)d_in[1];
    p.importance = (const float*)d_in[2];
    p.out        = (float*)d_out;
    p.tokens     = in_sizes[0] / 1024;

    p.eb0 = (const float*)d_in[EB + 0]; p.eb1 = (const float*)d_in[EB + 1]; p.eb2 = (const float*)d_in[EB + 2];
    p.eg0 = (const float*)d_in[EG + 0]; p.eg1 = (const float*)d_in[EG + 1]; p.eg2 = (const float*)d_in[EG + 2];
    p.et0 = (const float*)d_in[ET + 0]; p.et1 = (const float*)d_in[ET + 1]; p.et2 = (const float*)d_in[ET + 2];
    p.db0 = (const float*)d_in[DB + 0]; p.db1 = (const float*)d_in[DB + 1]; p.db2 = (const float*)d_in[DB + 2];
    p.dg0 = (const float*)d_in[DG + 0]; p.dg1 = (const float*)d_in[DG + 1]; p.dg2 = (const float*)d_in[DG + 2];
    p.dt0 = (const float*)d_in[DT + 0]; p.dt1 = (const float*)d_in[DT + 1]; p.dt2 = (const float*)d_in[DT + 2];

    // Pack six weight matrices (fp32 -> f16 WMMA-B swizzle) into workspace.
    size_t off = 0;
    auto pack = [&](const void* src, int Nr, int Kr, int Np, int Kp) -> const _Float16* {
        _Float16* dst = (_Float16*)d_ws + off;
        int total = Np * Kp;
        pikv_pack_w<<<(total + 255) / 256, 256, 0, stream>>>((const float*)src, dst, Nr, Kr, Np, Kp);
        off += (size_t)total;
        return dst;
    };
    p.We0 = pack(d_in[EW + 0],  512, 1024,  512, 1024);
    p.We1 = pack(d_in[EW + 1],  204,  512,  224,  512);
    p.We2 = pack(d_in[EW + 2],   65,  204,   96,  224);
    p.Wd2 = pack(d_in[DW + 2],  204,   65,  224,   96);
    p.Wd1 = pack(d_in[DW + 1],  512,  204,  512,  224);
    p.Wd0 = pack(d_in[DW + 0], 1024,  512, 1024,  512);

    const int tiles = 2 * (p.tokens / MROWS);   // keys + values
    pikv_main<<<tiles, 512, LDS_BYTES, stream>>>(p);
}
